// FactorizedVectorQuantize_17282948399510
// MI455X (gfx1250) — compile-verified, matrix operand order checked by reading, not run on hardware
//
#include <hip/hip_runtime.h>
#include <stdint.h>

typedef __attribute__((ext_vector_type(16))) _Float16 v16h;
typedef __attribute__((ext_vector_type(8)))  float    v8f;

#define B_SZ   8
#define D_IN   1024
#define T_SZ   2048
#define NTOK   (B_SZ * T_SZ)      // 16384 tokens
#define CB_N   8192
#define CB_D   8

#define QUARTERS    4
#define CODES_PER_Q (CB_N / QUARTERS)      // 2048
#define TILES_PER_Q (CODES_PER_Q / 16)     // 128

// d_out flat layout (4B elements), reference return order:
// out [B,1024,T] | commit[B] | cbloss[B] | indices[B,T] (int32) | z_e [B,8,T]
#define OUT_OFF     0
#define COMMIT_OFF  (B_SZ * D_IN * T_SZ)          // 16777216
#define CBLOSS_OFF  (COMMIT_OFF + B_SZ)
#define IDX_OFF     (CBLOSS_OFF + B_SZ)
#define ZE_OFF      (IDX_OFF + NTOK)

// workspace layout (bytes)
#define WS_CBPACK_OFF 0                            // 512 tiles * 512B (B-layout incl. zero half) = 256KB
#define WS_ENC_OFF    (512 * 512)                  // 16384*8 f16 = 256KB
#define WS_PVAL_OFF   (WS_ENC_OFF + NTOK * CB_D * 2)   // 16384*4 f32 = 256KB
#define WS_PIDX_OFF   (WS_PVAL_OFF + NTOK * QUARTERS * 4)

// ---------------------------------------------------------------------------
// Kernel 1: normalize codebook -> f16, pre-packed in WMMA B-matrix layout:
// per 16-code tile: 32 x 16B slots; lanes 0-15 hold code K0..7, lanes 16-31
// hold zeros (K8..15, K24..31 rows). Also zero the loss outputs.
// ---------------------------------------------------------------------------
__global__ __launch_bounds__(256) void prep_kernel(const float* __restrict__ codebook,
                                                   uint4* __restrict__ cb_pack,
                                                   float* __restrict__ losses) {
    if (blockIdx.x == 0 && threadIdx.x < 16) losses[threadIdx.x] = 0.0f;
    int i = blockIdx.x * 256 + threadIdx.x;
    if (i < CB_N) {
        const float4* cp = (const float4*)(codebook + (size_t)i * 8);
        float4 a = cp[0], b = cp[1];
        float v[8] = {a.x, a.y, a.z, a.w, b.x, b.y, b.z, b.w};
        float s = 0.0f;
#pragma unroll
        for (int k = 0; k < 8; ++k) s += v[k] * v[k];
        float inv = 1.0f / fmaxf(sqrtf(s), 1e-12f);
        union { _Float16 h[8]; uint4 q; } pk;
#pragma unroll
        for (int k = 0; k < 8; ++k) pk.h[k] = (_Float16)(v[k] * inv);
        int tile = i >> 4, lane = i & 15;
        uint4* dst = cb_pack + (size_t)tile * 32;
        uint4 zz = {0u, 0u, 0u, 0u};
        dst[lane]      = pk.q;   // lanes 0-15: K0..7 of code
        dst[16 + lane] = zz;     // lanes 16-31: zero K rows
    }
}

// ---------------------------------------------------------------------------
// Kernel 2: z_e = w_in @ z + b_in (one thread per token column), write z_e
// (fp32 output) and L2-normalized enc as f16 to workspace. 64MB-read bound.
// ---------------------------------------------------------------------------
__global__ __launch_bounds__(256) void project_kernel(const float* __restrict__ z,
                                                      const float* __restrict__ w_in,
                                                      const float* __restrict__ b_in,
                                                      float* __restrict__ ze_out,
                                                      _Float16* __restrict__ enc_h) {
    __shared__ float s_win[D_IN * 8];   // [d*8 + o]
    for (int j = threadIdx.x; j < D_IN * 8; j += 256) {
        int d = j >> 3, o = j & 7;
        s_win[j] = w_in[o * D_IN + d];
    }
    __syncthreads();

    int g = blockIdx.x * 256 + threadIdx.x;
    int b = g / T_SZ, t = g % T_SZ;
    float acc[8];
#pragma unroll
    for (int o = 0; o < 8; ++o) acc[o] = b_in[o];

    const float*  zp = z + (size_t)b * D_IN * T_SZ + t;
    const float4* w4 = (const float4*)s_win;
#pragma unroll 4
    for (int d = 0; d < D_IN; ++d) {
        float  zv = zp[(size_t)d * T_SZ];
        float4 w0 = w4[d * 2], w1 = w4[d * 2 + 1];
        acc[0] += w0.x * zv; acc[1] += w0.y * zv; acc[2] += w0.z * zv; acc[3] += w0.w * zv;
        acc[4] += w1.x * zv; acc[5] += w1.y * zv; acc[6] += w1.z * zv; acc[7] += w1.w * zv;
    }

    float s = 0.0f;
#pragma unroll
    for (int o = 0; o < 8; ++o) s += acc[o] * acc[o];
    float inv = 1.0f / fmaxf(sqrtf(s), 1e-12f);

#pragma unroll
    for (int o = 0; o < 8; ++o)
        ze_out[((size_t)b * 8 + o) * T_SZ + t] = acc[o];

    union { _Float16 h[8]; uint4 q; } pk;
#pragma unroll
    for (int o = 0; o < 8; ++o) pk.h[o] = (_Float16)(acc[o] * inv);
    *(uint4*)(enc_h + (size_t)g * 8) = pk.q;
}

// ---------------------------------------------------------------------------
// Kernel 3 (WMMA): block = 8 waves, all working on one codebook quarter.
// The 64KB pre-packed B slice is staged in LDS once; inner loop is one
// unpredicated ds_load_b128 + one v_wmma_f32_16x16x32_f16 + argmax update.
// Unroll 4 gives the scheduler independent work to fill the WMMA->VALU
// hazard slots (5-cycle RAW on D). Each wave: 16 tokens x 2048 codes.
// ---------------------------------------------------------------------------
__global__ __launch_bounds__(256) void argmin_kernel(const _Float16* __restrict__ enc_h,
                                                     const uint4* __restrict__ cb_pack,
                                                     float* __restrict__ pval,
                                                     int* __restrict__ pidx) {
    __shared__ uint4 sB[TILES_PER_Q * 32];          // 64 KB

    int quarter  = blockIdx.x & (QUARTERS - 1);
    int tokGroup = blockIdx.x >> 2;                 // 0..127
    const uint4* src = cb_pack + (size_t)quarter * TILES_PER_Q * 32;
    for (int j = threadIdx.x; j < TILES_PER_Q * 32; j += 256) sB[j] = src[j];
    __syncthreads();

    int lane   = threadIdx.x & 31;
    int waveId = threadIdx.x >> 5;
    int base   = tokGroup * 128 + waveId * 16;      // first token of this wave's tile

    union HV { v16h v; unsigned int u[8]; };
    HV A;
#pragma unroll
    for (int k = 0; k < 8; ++k) A.u[k] = 0u;
    if (lane < 16) {                                // one-time predicated A load
        uint4 q = ((const uint4*)enc_h)[base + lane];
        A.u[0] = q.x; A.u[1] = q.y; A.u[2] = q.z; A.u[3] = q.w;
    }

    HV Bv;                                          // upper half stays zero forever
#pragma unroll
    for (int k = 0; k < 8; ++k) Bv.u[k] = 0u;

    float best[8];
    int   bidx[8];
#pragma unroll
    for (int r = 0; r < 8; ++r) { best[r] = -3.0e38f; bidx[r] = 0; }

    int code = quarter * CODES_PER_Q + (lane & 15);
#pragma unroll 4
    for (int tile = 0; tile < TILES_PER_Q; ++tile) {
        uint4 q = sB[tile * 32 + lane];             // all lanes active, no exec dance
        Bv.u[0] = q.x; Bv.u[1] = q.y; Bv.u[2] = q.z; Bv.u[3] = q.w;
        v8f c = {};
        v8f d = __builtin_amdgcn_wmma_f32_16x16x32_f16(false, A.v, false, Bv.v,
                                                       (short)0, c, false, false);
#pragma unroll
        for (int r = 0; r < 8; ++r) {
            float dv = d[r];
            bool  gt = dv > best[r];                // strict >: first (lowest) index wins ties
            best[r] = gt ? dv   : best[r];
            bidx[r] = gt ? code : bidx[r];
        }
        code += 16;
    }

    // argmax reduction over the 16 lanes (N dim) of each half-wave
#pragma unroll
    for (int m = 1; m < 16; m <<= 1) {
#pragma unroll
        for (int r = 0; r < 8; ++r) {
            float ov = __shfl_xor(best[r], m, 16);
            int   oi = __shfl_xor(bidx[r], m, 16);
            bool take = (ov > best[r]) || (ov == best[r] && oi < bidx[r]);
            best[r] = take ? ov : best[r];
            bidx[r] = take ? oi : bidx[r];
        }
    }

    if ((lane & 15) == 0) {
        int half = lane >> 4;                       // 0: M=r, 1: M=r+8
#pragma unroll
        for (int r = 0; r < 8; ++r) {
            int token = base + half * 8 + r;
            pval[token * QUARTERS + quarter] = best[r];
            pidx[token * QUARTERS + quarter] = bidx[r];
        }
    }
}

// ---------------------------------------------------------------------------
// Kernel 3b: merge the 4 per-quarter candidates per token.
// ---------------------------------------------------------------------------
__global__ __launch_bounds__(256) void combine_kernel(const float* __restrict__ pval,
                                                      const int* __restrict__ pidx,
                                                      int* __restrict__ idx_out) {
    int g = blockIdx.x * 256 + threadIdx.x;
    float bv = pval[g * QUARTERS];
    int   bi = pidx[g * QUARTERS];
#pragma unroll
    for (int q = 1; q < QUARTERS; ++q) {
        float v = pval[g * QUARTERS + q];
        int   i = pidx[g * QUARTERS + q];
        bool take = v > bv;                         // quarters scanned in ascending code order
        bv = take ? v : bv;
        bi = take ? i : bi;
    }
    idx_out[g] = bi;
}

// ---------------------------------------------------------------------------
// Kernel 4: out = w_out @ codebook[idx] + b_out. 64MB-store bound.
// ---------------------------------------------------------------------------
__global__ __launch_bounds__(256) void output_kernel(const float* __restrict__ codebook,
                                                     const float* __restrict__ w_out,
                                                     const float* __restrict__ b_out,
                                                     const int* __restrict__ idx_in,
                                                     float* __restrict__ out) {
    __shared__ float s_wout[D_IN * 8];
    __shared__ float s_bout[D_IN];
    for (int j = threadIdx.x; j < D_IN * 8; j += 256) s_wout[j] = w_out[j];
    for (int j = threadIdx.x; j < D_IN; j += 256)     s_bout[j] = b_out[j];
    __syncthreads();

    int g = blockIdx.x * 256 + threadIdx.x;
    int b = g / T_SZ, t = g % T_SZ;
    int idx = idx_in[g];

    const float4* cp = (const float4*)(codebook + (size_t)idx * 8);
    float4 q0 = cp[0], q1 = cp[1];

    const float4* w4 = (const float4*)s_wout;
    float* op = out + (size_t)b * D_IN * T_SZ + t;
#pragma unroll 4
    for (int o = 0; o < D_IN; ++o) {
        float4 w0 = w4[o * 2], w1 = w4[o * 2 + 1];
        float acc = s_bout[o];
        acc += w0.x * q0.x + w0.y * q0.y + w0.z * q0.z + w0.w * q0.w;
        acc += w1.x * q1.x + w1.y * q1.y + w1.z * q1.z + w1.w * q1.w;
        op[(size_t)o * T_SZ] = acc;
    }
}

// ---------------------------------------------------------------------------
extern "C" void kernel_launch(void* const* d_in, const int* in_sizes, int n_in,
                              void* d_out, int out_size, void* d_ws, size_t ws_size,
                              hipStream_t stream) {
    const float* z        = (const float*)d_in[0];
    const float* w_in     = (const float*)d_in[1];
    const float* b_in     = (const float*)d_in[2];
    const float* w_out    = (const float*)d_in[3];
    const float* b_out    = (const float*)d_in[4];
    const float* codebook = (const float*)d_in[5];

    float*    outf    = (float*)d_out;
    float*    losses  = outf + COMMIT_OFF;
    int*      idxs    = (int*)(outf + IDX_OFF);
    float*    ze      = outf + ZE_OFF;
    uint4*    cb_pack = (uint4*)((char*)d_ws + WS_CBPACK_OFF);
    _Float16* enc_h   = (_Float16*)((char*)d_ws + WS_ENC_OFF);
    float*    pval    = (float*)((char*)d_ws + WS_PVAL_OFF);
    int*      pidx    = (int*)((char*)d_ws + WS_PIDX_OFF);

    prep_kernel   <<<CB_N / 256, 256, 0, stream>>>(codebook, cb_pack, losses);
    project_kernel<<<NTOK / 256, 256, 0, stream>>>(z, w_in, b_in, ze, enc_h);
    argmin_kernel <<<(NTOK / 128) * QUARTERS, 256, 0, stream>>>(enc_h, cb_pack, pval, pidx);
    combine_kernel<<<NTOK / 256, 256, 0, stream>>>(pval, pidx, idxs);
    output_kernel <<<NTOK / 256, 256, 0, stream>>>(codebook, w_out, b_out, idxs, outf + OUT_OFF);
}
